// CroAttention_42236708389243
// MI455X (gfx1250) — matrix operand order repaired and checked by reading, not compiled, not run on hardware
//
#include <hip/hip_runtime.h>
#include <hip/hip_bf16.h>

// ---- problem constants ----
#define BB 4
#define CC 512
#define TT 32
#define LL 256
#define HH 8
#define DD 64
#define BT (BB*TT)

typedef __bf16 bf16_t;
typedef __attribute__((ext_vector_type(16))) __bf16    v16bf;
typedef __attribute__((ext_vector_type(8)))  float     v8f;
typedef __attribute__((ext_vector_type(4)))  unsigned  v4u;
typedef __attribute__((ext_vector_type(8)))  unsigned  v8u;

__device__ __forceinline__ v8f wmma_bf16(v16bf a, v16bf b, v8f c) {
  // D = A(16x32) * B(32x16) + C, f32 accumulate
  return __builtin_amdgcn_wmma_f32_16x16x32_bf16(false, a, false, b, (short)0, c,
                                                 false, false);
}

// A fragment (16x32 bf16, row-major LDS tile, 'stride' elements per row).
// Lane < 16: M = lane, K = 0..15; lane >= 16: M = lane-16, K = 16..31.
__device__ __forceinline__ v16bf load_frag_a(const bf16_t* base, int stride, int lane) {
  const int m  = lane & 15;
  const int kh = (lane >> 4) << 4;
  const bf16_t* p = base + m * stride;
  v16bf r;
#pragma unroll
  for (int j = 0; j < 16; ++j) r[j] = p[kh + j];
  return r;
}

// B fragment (32x16, stored transposed in LDS as [n][k] row-major).
// Lane < 16: N = lane, K = 0..15; lane >= 16: N = lane-16, K = 16..31.
__device__ __forceinline__ v16bf load_frag_b(const bf16_t* base, int stride, int lane) {
  const int n  = lane & 15;
  const int kh = (lane >> 4) << 4;
  const bf16_t* p = base + n * stride + kh;
  v16bf r;
#pragma unroll
  for (int j = 0; j < 16; ++j) r[j] = p[j];
  return r;
}

__device__ __forceinline__ unsigned mix32(unsigned x) {
  x ^= x >> 16; x *= 0x7feb352dU;
  x ^= x >> 15; x *= 0x846ca68bU;
  x ^= x >> 16; return x;
}

// ---------------------------------------------------------------------------
// Tensor Data Mover: 1-row (contiguous) global -> LDS copy of `nelem` bf16
// elements. D# per CDNA5 ISA:
//   group0: count=1, lds_addr, 57-bit global addr, type=2
//   group1: data_size=1 (2B), tensor_dim0=nelem, tensor_dim1=1,
//           tile_dim0=nelem (<=65535), tile_dim1=1, dim0_stride=nelem
// Wave-level DMA (EXEC ignored); issue from one wave, complete with
// s_wait_tensorcnt, then publish via workgroup barrier.
// ---------------------------------------------------------------------------
__device__ __forceinline__ void tdm_load_1d(unsigned lds_byte_off,
                                            const void* gsrc, unsigned nelem) {
  unsigned long long ga = (unsigned long long)(uintptr_t)gsrc;
  v4u g0;
  g0[0] = 1u;                                             // count=1, user mode
  g0[1] = lds_byte_off;                                   // lds_addr
  g0[2] = (unsigned)ga;                                   // global_addr[31:0]
  g0[3] = (unsigned)((ga >> 32) & 0x1FFFFFFu) | 0x80000000u; // addr[56:32]|type=2
  v8u g1;
  g1[0] = 0x00010000u;                    // data_size=1 (2 bytes), no multicast
  g1[1] = (nelem & 0xFFFFu) << 16;        // tensor_dim0[15:0]
  g1[2] = ((nelem >> 16) & 0xFFFFu) | (1u << 16); // dim0[31:16] | tensor_dim1=1
  g1[3] = (nelem & 0xFFFFu) << 16;        // tile_dim0
  g1[4] = 1u;                             // tile_dim1=1, tile_dim2=0
  g1[5] = nelem;                          // tensor_dim0_stride[31:0]
  g1[6] = 0u;                             // stride hi | tensor_dim1_stride lo
  g1[7] = 0u;
  asm volatile("tensor_load_to_lds %0, %1" :: "s"(g0), "s"(g1) : "memory");
}

// ---------------------------------------------------------------------------
// Kernel 0: f32 -> bf16 elementwise (weights)
// ---------------------------------------------------------------------------
__global__ void f32_to_bf16_kernel(const float* __restrict__ s,
                                   bf16_t* __restrict__ d, int n) {
  int i = blockIdx.x * 256 + threadIdx.x;
  if (i < n) d[i] = (bf16_t)s[i];
}

// ---------------------------------------------------------------------------
// Kernel 1: projection + bias + per-head L2 norm.
//   in  : f32 tensor [B, C, T, L]  (read as A[l, c] = in[b, c, t, l])
//   W   : bf16 [Ntot, C] (row = out channel, contiguous K)
//   out : bf16 [bt, head, l, d]; heads < boundary -> out0, else out1
// Block: 256 thr, tile M=128 (l) x N=64 (one head), K-loop over C.
// ---------------------------------------------------------------------------
__global__ __launch_bounds__(256) void proj_norm_kernel(
    const float* __restrict__ in, const bf16_t* __restrict__ W,
    const float* __restrict__ bias, bf16_t* __restrict__ out0,
    bf16_t* __restrict__ out1, int boundary) {
  const int bt = blockIdx.x;
  const int b  = bt / TT, t = bt % TT;
  const int h  = blockIdx.y;
  const int l0 = blockIdx.z * 128;

  __shared__ bf16_t Asm[128 * 40];
  __shared__ bf16_t Bsm[64 * 40];
  __shared__ float  Osm[128 * 68];
  __shared__ float  rscale[128];

  const int tid  = threadIdx.x;
  const int lane = tid & 31, wid = tid >> 5;
  const int wm = wid & 3, wn = wid >> 2;      // 4 (M) x 2 (N) wave grid
  const int half = lane >> 4, nl = lane & 15;
  const int nBase = h * 64;

  v8f acc[2][2] = {};

  for (int k0 = 0; k0 < CC; k0 += 32) {
    // prefetch next A tile (global_prefetch_b8)
    if (k0 + 32 < CC)
      __builtin_prefetch(
          &in[(((size_t)b * CC + (k0 + 32 + (tid >> 7))) * TT + t) * LL + l0 +
              (tid & 127)], 0, 0);
    // A tile 128x32: in[b, k0+k, t, l0+m], coalesced along l.
#pragma unroll
    for (int i = 0; i < 16; ++i) {
      int idx = tid + i * 256;
      int k = idx >> 7, m = idx & 127;
      float v = in[(((size_t)b * CC + (k0 + k)) * TT + t) * LL + l0 + m];
      Asm[m * 40 + k] = (bf16_t)v;
    }
    // W tile 64x32, vectorized 16B copies.
    {
      int n = tid >> 2, q4 = tid & 3;
      const uint4* src = (const uint4*)(W + (size_t)(nBase + n) * CC + k0);
      ((uint4*)(&Bsm[n * 40]))[q4] = src[q4];
    }
    __syncthreads();

    v16bf bf0 = load_frag_b(&Bsm[(wn * 32 + 0)  * 40], 40, lane);
    v16bf bf1 = load_frag_b(&Bsm[(wn * 32 + 16) * 40], 40, lane);
#pragma unroll
    for (int mf = 0; mf < 2; ++mf) {
      v16bf af = load_frag_a(&Asm[(wm * 32 + mf * 16) * 40], 40, lane);
      acc[mf][0] = wmma_bf16(af, bf0, acc[mf][0]);
      acc[mf][1] = wmma_bf16(af, bf1, acc[mf][1]);
    }
    __syncthreads();
  }

  // bias + stage to LDS for cross-wave row L2 norm
#pragma unroll
  for (int mf = 0; mf < 2; ++mf)
#pragma unroll
    for (int nf = 0; nf < 2; ++nf) {
      const int cn = wn * 32 + nf * 16 + nl;
      const float bb = bias[nBase + cn];
#pragma unroll
      for (int r = 0; r < 8; ++r) {
        const int m = wm * 32 + mf * 16 + half * 8 + r;
        Osm[m * 68 + cn] = acc[mf][nf][r] + bb;
      }
    }
  __syncthreads();

  if (tid < 128) {
    float ss = 0.f;
    for (int d = 0; d < 64; ++d) { float v = Osm[tid * 68 + d]; ss += v * v; }
    rscale[tid] = 1.0f / fmaxf(sqrtf(ss), 1e-12f);
  }
  __syncthreads();

  bf16_t* outp; int hh;
  if (h < boundary) { outp = out0; hh = h; } else { outp = out1; hh = h - boundary; }
  const size_t obase = ((size_t)(bt * 8 + hh) * LL + l0) * DD;
  for (int i = 0; i < 32; ++i) {
    int idx = tid + i * 256;       // 128*64
    int m = idx >> 6, d = idx & 63;
    outp[obase + (size_t)m * DD + d] = (bf16_t)(Osm[m * 68 + d] * rscale[m]);
  }
}

// ---------------------------------------------------------------------------
// Kernel 2: flash attention per (bt, head). q,k,v bf16 [bt,h,l,d].
// 8 waves; wave owns 32 query rows. q tile + k blocks come in via the
// Tensor Data Mover (tensor_load_to_lds + s_wait_tensorcnt). Stream 32-key
// blocks: S = q k^T (wmma), hash dropout + 1/sqrt(D), online softmax,
// O += p v (wmma, p relaid out via per-wave LDS).
// ---------------------------------------------------------------------------
__global__ __launch_bounds__(256) void attn_kernel(
    const bf16_t* __restrict__ q, const bf16_t* __restrict__ k,
    const bf16_t* __restrict__ v, bf16_t* __restrict__ o) {
  const int bt = blockIdx.x, h = blockIdx.y;
  const int tid = threadIdx.x, lane = tid & 31, wid = tid >> 5;
  const int half = lane >> 4, nl = lane & 15;

  __shared__ bf16_t qsm[256 * 64];   // full q tile
  __shared__ bf16_t ksm[32 * 64];    // key block, [m][d]
  __shared__ bf16_t vsmT[64 * 40];   // value block transposed, [d][m]
  __shared__ bf16_t psm[8 * 32 * 40];// per-wave p scratch

  const size_t qb = (size_t)(bt * HH + h) * (LL * DD);
  const unsigned qsm_off = (unsigned)(uintptr_t)&qsm[0]; // LDS addr = low 32 bits
  const unsigned ksm_off = (unsigned)(uintptr_t)&ksm[0];

  // q tile (32 KB) via TDM from wave 0
  if (wid == 0) {
    tdm_load_1d(qsm_off, q + qb, LL * DD);
    __builtin_amdgcn_s_wait_tensorcnt(0);
  }
  __syncthreads();

  // resident q A-fragments: [mf][kstep]
  v16bf aq[2][2];
#pragma unroll
  for (int mf = 0; mf < 2; ++mf)
#pragma unroll
    for (int ks = 0; ks < 2; ++ks)
      aq[mf][ks] = load_frag_a(&qsm[(wid * 32 + mf * 16) * 64 + ks * 32], 64, lane);

  v8f rmax[2], rsum[2], oacc[2][4];
#pragma unroll
  for (int mf = 0; mf < 2; ++mf) {
#pragma unroll
    for (int r = 0; r < 8; ++r) { rmax[mf][r] = -1e30f; rsum[mf][r] = 0.f; }
#pragma unroll
    for (int nf = 0; nf < 4; ++nf)
#pragma unroll
      for (int r = 0; r < 8; ++r) oacc[mf][nf][r] = 0.f;
  }

  bf16_t* pw = &psm[wid * 32 * 40];
  const unsigned hbase = (unsigned)(bt * HH + h) * 256u;

  for (int nb = 0; nb < 8; ++nb) {
    const int n0 = nb * 32;
    __syncthreads();                       // previous block's reads complete
    // key block [m][d] via TDM (4 KB contiguous)
    if (wid == 0) tdm_load_1d(ksm_off, k + qb + (size_t)n0 * DD, 32 * DD);
    // value block transposed [d][m] (manual: TDM cannot transpose)
    {
      const bf16_t* vsrc = v + qb + (size_t)n0 * DD;
#pragma unroll
      for (int i = 0; i < 8; ++i) {
        int idx = tid + i * 256;
        int m = idx >> 6, d = idx & 63;
        vsmT[d * 40 + m] = vsrc[m * 64 + d];
      }
    }
    if (wid == 0) __builtin_amdgcn_s_wait_tensorcnt(0);
    __syncthreads();

    // S = q k^T  (B frags from ksm rows = key index)
    v8f s[2][2] = {};
#pragma unroll
    for (int ks = 0; ks < 2; ++ks) {
      v16bf bk0 = load_frag_b(&ksm[0  * 64 + ks * 32], 64, lane);
      v16bf bk1 = load_frag_b(&ksm[16 * 64 + ks * 32], 64, lane);
#pragma unroll
      for (int mf = 0; mf < 2; ++mf) {
        s[mf][0] = wmma_bf16(aq[mf][ks], bk0, s[mf][0]);
        s[mf][1] = wmma_bf16(aq[mf][ks], bk1, s[mf][1]);
      }
    }

    // dropout (before softmax) + 1/sqrt(D)
#pragma unroll
    for (int mf = 0; mf < 2; ++mf)
#pragma unroll
      for (int nf = 0; nf < 2; ++nf)
#pragma unroll
        for (int r = 0; r < 8; ++r) {
          const int row = wid * 32 + mf * 16 + half * 8 + r;
          const int col = n0 + nf * 16 + nl;
          unsigned idx = ((hbase + (unsigned)row) << 8) | (unsigned)col;
          unsigned hv = mix32(idx * 0x9E3779B9u + 42u);
          float sv = s[mf][nf][r];
          sv = (hv < 3865470566u) ? sv * (1.0f / 0.9f) : 0.0f;
          s[mf][nf][r] = sv * 0.125f;
        }

    // online softmax update
#pragma unroll
    for (int mf = 0; mf < 2; ++mf) {
      v8f bmax;
#pragma unroll
      for (int r = 0; r < 8; ++r) bmax[r] = fmaxf(s[mf][0][r], s[mf][1][r]);
#pragma unroll
      for (int off = 8; off >= 1; off >>= 1)
#pragma unroll
        for (int r = 0; r < 8; ++r)
          bmax[r] = fmaxf(bmax[r], __shfl_xor(bmax[r], off, 32));
#pragma unroll
      for (int r = 0; r < 8; ++r) {
        float mo = rmax[mf][r];
        float mn = fmaxf(mo, bmax[r]);
        float sc = __expf(mo - mn);
        rmax[mf][r] = mn;
        rsum[mf][r] *= sc;
#pragma unroll
        for (int nf = 0; nf < 4; ++nf) oacc[mf][nf][r] *= sc;
      }
#pragma unroll
      for (int nf = 0; nf < 2; ++nf)
#pragma unroll
        for (int r = 0; r < 8; ++r)
          s[mf][nf][r] = __expf(s[mf][nf][r] - rmax[mf][r]);
      v8f bs;
#pragma unroll
      for (int r = 0; r < 8; ++r) bs[r] = s[mf][0][r] + s[mf][1][r];
#pragma unroll
      for (int off = 8; off >= 1; off >>= 1)
#pragma unroll
        for (int r = 0; r < 8; ++r) bs[r] += __shfl_xor(bs[r], off, 32);
#pragma unroll
      for (int r = 0; r < 8; ++r) rsum[mf][r] += bs[r];

      // spill p (D-layout) to per-wave LDS so it can be re-read as A fragments
#pragma unroll
      for (int nf = 0; nf < 2; ++nf)
#pragma unroll
        for (int r = 0; r < 8; ++r) {
          const int rl = mf * 16 + half * 8 + r;
          const int cl = nf * 16 + nl;
          pw[rl * 40 + cl] = (bf16_t)s[mf][nf][r];
        }
    }

    // O += p (32x32) * v (32x64); B frags from transposed v tile
#pragma unroll
    for (int mf = 0; mf < 2; ++mf) {
      v16bf ap = load_frag_a(&pw[(mf * 16) * 40], 40, lane);
#pragma unroll
      for (int nf = 0; nf < 4; ++nf) {
        v16bf bv = load_frag_b(&vsmT[(nf * 16) * 40], 40, lane);
        oacc[mf][nf] = wmma_bf16(ap, bv, oacc[mf][nf]);
      }
    }
  }

  // normalize and store o[bt, l, h*64 + d] (bf16, [bt,l,C] layout)
#pragma unroll
  for (int mf = 0; mf < 2; ++mf)
#pragma unroll
    for (int nf = 0; nf < 4; ++nf) {
      const int d = nf * 16 + nl;
#pragma unroll
      for (int r = 0; r < 8; ++r) {
        const int l = wid * 32 + mf * 16 + half * 8 + r;
        o[((size_t)bt * LL + l) * CC + h * DD + d] =
            (bf16_t)(oacc[mf][nf][r] / rsum[mf][r]);
      }
    }
}

// ---------------------------------------------------------------------------
// Kernel 3: out projection + bias + residual, transposed f32 store.
//   o  : bf16 [bt, l, c']   Wm : bf16 [c, c']   y[b,c,t,l] = o Wm^T + bm + x
// ---------------------------------------------------------------------------
__global__ __launch_bounds__(256) void outproj_kernel(
    const bf16_t* __restrict__ o, const bf16_t* __restrict__ Wm,
    const float* __restrict__ bm, const float* __restrict__ xin,
    float* __restrict__ y) {
  const int bt = blockIdx.x;
  const int b  = bt / TT, t = bt % TT;
  const int n0 = blockIdx.y * 64;
  const int l0 = blockIdx.z * 128;

  __shared__ bf16_t Asm[128 * 40];
  __shared__ bf16_t Bsm[64 * 40];

  const int tid = threadIdx.x, lane = tid & 31, wid = tid >> 5;
  const int wm = wid & 3, wn = wid >> 2;
  const int half = lane >> 4, nl = lane & 15;

  v8f acc[2][2] = {};

  for (int k0 = 0; k0 < CC; k0 += 32) {
    if (k0 + 32 < CC)
      __builtin_prefetch(o + ((size_t)bt * LL + l0 + (tid >> 1)) * CC + (k0 + 32),
                         0, 0);
    // A tile 128x32 from o (contiguous K), 16B vector copies: 512 uint4
    {
#pragma unroll
      for (int i = 0; i < 2; ++i) {
        int idx = tid + i * 256;
        int m = idx >> 2, q4 = idx & 3;
        const uint4* src =
            (const uint4*)(o + ((size_t)bt * LL + l0 + m) * CC + k0);
        ((uint4*)(&Asm[m * 40]))[q4] = src[q4];
      }
    }
    // B tile 64x32 from Wm rows n0..n0+63
    {
      int n = tid >> 2, q4 = tid & 3;
      const uint4* src = (const uint4*)(Wm + (size_t)(n0 + n) * CC + k0);
      ((uint4*)(&Bsm[n * 40]))[q4] = src[q4];
    }
    __syncthreads();

    v16bf bf0 = load_frag_b(&Bsm[(wn * 32 + 0)  * 40], 40, lane);
    v16bf bf1 = load_frag_b(&Bsm[(wn * 32 + 16) * 40], 40, lane);
#pragma unroll
    for (int mf = 0; mf < 2; ++mf) {
      v16bf af = load_frag_a(&Asm[(wm * 32 + mf * 16) * 40], 40, lane);
      acc[mf][0] = wmma_bf16(af, bf0, acc[mf][0]);
      acc[mf][1] = wmma_bf16(af, bf1, acc[mf][1]);
    }
    __syncthreads();
  }

#pragma unroll
  for (int mf = 0; mf < 2; ++mf)
#pragma unroll
    for (int nf = 0; nf < 2; ++nf) {
      const int c  = n0 + wn * 32 + nf * 16 + nl;
      const int lb = l0 + wm * 32 + mf * 16 + half * 8;
      const size_t base = (((size_t)b * CC + c) * TT + t) * LL + lb;
      const float bmc = bm[c];
#pragma unroll
      for (int r = 0; r < 8; ++r)
        y[base + r] = acc[mf][nf][r] + bmc + xin[base + r];
    }
}

// ---------------------------------------------------------------------------
extern "C" void kernel_launch(void* const* d_in, const int* in_sizes, int n_in,
                              void* d_out, int out_size, void* d_ws, size_t ws_size,
                              hipStream_t stream) {
  (void)in_sizes; (void)n_in; (void)out_size; (void)ws_size;
  const float* e   = (const float*)d_in[0];
  const float* x   = (const float*)d_in[1];
  const float* Wq  = (const float*)d_in[2];
  const float* bq  = (const float*)d_in[3];
  const float* Wkv = (const float*)d_in[4];
  const float* bkv = (const float*)d_in[5];
  const float* Wm  = (const float*)d_in[6];
  const float* bm  = (const float*)d_in[7];
  float* y = (float*)d_out;

  char* ws = (char*)d_ws;
  const size_t szWq  = (size_t)CC * CC * 2;
  const size_t szWkv = (size_t)2 * CC * CC * 2;
  const size_t szQKV = (size_t)BT * HH * LL * DD * 2;   // 32 MB each
  bf16_t* wq_bf  = (bf16_t*)(ws);
  bf16_t* wkv_bf = (bf16_t*)(ws + szWq);
  bf16_t* wm_bf  = (bf16_t*)(ws + szWq + szWkv);
  bf16_t* q_bf   = (bf16_t*)(ws + szWq + szWkv + szWq);
  bf16_t* k_bf   = (bf16_t*)((char*)q_bf + szQKV);
  bf16_t* v_bf   = (bf16_t*)((char*)k_bf + szQKV);
  bf16_t* o_bf   = (bf16_t*)((char*)v_bf + szQKV);

  dim3 blk(256);
  const int nWq = CC * CC, nWkv = 2 * CC * CC;
  f32_to_bf16_kernel<<<(nWq  + 255) / 256, blk, 0, stream>>>(Wq,  wq_bf,  nWq);
  f32_to_bf16_kernel<<<(nWkv + 255) / 256, blk, 0, stream>>>(Wkv, wkv_bf, nWkv);
  f32_to_bf16_kernel<<<(nWq  + 255) / 256, blk, 0, stream>>>(Wm,  wm_bf,  nWq);

  proj_norm_kernel<<<dim3(BT, HH, 2), blk, 0, stream>>>(
      e, wq_bf, bq, q_bf, q_bf, HH);
  proj_norm_kernel<<<dim3(BT, 2 * HH, 2), blk, 0, stream>>>(
      x, wkv_bf, bkv, k_bf, v_bf, HH);

  attn_kernel<<<dim3(BT, HH), blk, 0, stream>>>(q_bf, k_bf, v_bf, o_bf);

  outproj_kernel<<<dim3(BT, CC / 64, 2), blk, 0, stream>>>(
      o_bf, wm_bf, bm, x, y);
}
